// ScaledDotProductAttention_50706383896798
// MI455X (gfx1250) — matrix operand order repaired and checked by reading, not compiled, not run on hardware
//
#include <hip/hip_runtime.h>
#include <hip/hip_bf16.h>

typedef __attribute__((ext_vector_type(16))) _Float16 v16h;
typedef __attribute__((ext_vector_type(8)))  _Float16 v8h;
typedef __attribute__((ext_vector_type(8)))  float    v8f;

constexpr int DMODEL = 1024;
constexpr int DK     = 64;
constexpr int NHEAD  = 16;
constexpr int NQ     = 2048;
constexpr int NK     = 2048;
constexpr int NBATCH = 2;

// ---------------------------------------------------------------------------
// WMMA wrapper: D = A(16x32 f16) * B(32x16 f16) + C(16x16 f32)
// ---------------------------------------------------------------------------
__device__ __forceinline__ v8f wmma_f16(v16h a, v16h b, v8f c) {
  return __builtin_amdgcn_wmma_f32_16x16x32_f16(
      /*neg_a=*/false, a, /*neg_b=*/false, b,
      /*c_mod=*/(short)0, c, /*reuse_a=*/false, /*reuse_b=*/false);
}

// ---------------------------------------------------------------------------
// Fragment loaders (layouts per CDNA5 ISA 7.12.2)
// A 16x32 f16: lane l (l16=l&15,hf=l>>4): row=l16, halves 0..7 -> K=hf*8+i,
//              halves 8..15 -> K=16+hf*8+i
// B 32x16 f16: lane l: col=l16, halves 0..7 -> K=hf*8+i,
//              halves 8..15 -> K=16+hf*8+i
// Storage convention: "base + r*ld + k" where r is the row (A) / column (B)
// index and k is contiguous in memory.
// ---------------------------------------------------------------------------
__device__ __forceinline__ v16h fragA_f32(const float* __restrict__ base,
                                          size_t ld, int r0, int k0) {
  const int lane = threadIdx.x & 31, l16 = lane & 15, hf = lane >> 4;
  const float* p = base + (size_t)(r0 + l16) * ld + (k0 + hf * 8);
  v16h f;
#pragma unroll
  for (int i = 0; i < 8; ++i) {
    f[i]     = (_Float16)p[i];
    f[i + 8] = (_Float16)p[i + 16];
  }
  return f;
}

__device__ __forceinline__ v16h fragA_f16(const _Float16* __restrict__ base,
                                          size_t ld, int r0, int k0) {
  const int lane = threadIdx.x & 31, l16 = lane & 15, hf = lane >> 4;
  const _Float16* p = base + (size_t)(r0 + l16) * ld + (k0 + hf * 8);
  v16h f;
#pragma unroll
  for (int i = 0; i < 8; ++i) {
    f[i]     = p[i];
    f[i + 8] = p[i + 16];
  }
  return f;
}

__device__ __forceinline__ v16h fragB_f32(const float* __restrict__ base,
                                          size_t ld, int n0, int k0) {
  const int lane = threadIdx.x & 31, l16 = lane & 15, hf = lane >> 4;
  const float* p = base + (size_t)(n0 + l16) * ld + (k0 + hf * 8);
  v16h f;
#pragma unroll
  for (int i = 0; i < 8; ++i) {
    f[i]     = (_Float16)p[i];
    f[i + 8] = (_Float16)p[i + 16];
  }
  return f;
}

__device__ __forceinline__ v16h fragB_f16(const _Float16* __restrict__ base,
                                          size_t ld, int n0, int k0) {
  const int lane = threadIdx.x & 31, l16 = lane & 15, hf = lane >> 4;
  const _Float16* p = base + (size_t)(n0 + l16) * ld + (k0 + hf * 8);
  v16h f;
#pragma unroll
  for (int i = 0; i < 8; ++i) {
    f[i]     = p[i];
    f[i + 8] = p[i + 16];
  }
  return f;
}

// 16-lane (half-wave) reductions: xor masks < 16 keep each half independent.
__device__ __forceinline__ float half16_max(float v) {
  v = fmaxf(v, __shfl_xor(v, 1, 32));
  v = fmaxf(v, __shfl_xor(v, 2, 32));
  v = fmaxf(v, __shfl_xor(v, 4, 32));
  v = fmaxf(v, __shfl_xor(v, 8, 32));
  return v;
}
__device__ __forceinline__ float half16_sum(float v) {
  v += __shfl_xor(v, 1, 32);
  v += __shfl_xor(v, 2, 32);
  v += __shfl_xor(v, 4, 32);
  v += __shfl_xor(v, 8, 32);
  return v;
}

// ---------------------------------------------------------------------------
// Kernel 1: projection GEMM  C[4096,1024] = X[4096,1024] @ W[1024,1024]^T + b
// One 16x16 output tile per wave, 8 waves/block. TRANSPOSED=false: f16 out
// as [B,H,N,64] (Q,K). TRANSPOSED=true: f16 out as [B,H,64,N] (V) with the
// 8 per-lane rows packed into a single b128 store (consecutive q).
// ---------------------------------------------------------------------------
template <bool TRANSPOSED>
__global__ __launch_bounds__(256) void proj_kernel(
    const float* __restrict__ X, const float* __restrict__ W,
    const float* __restrict__ bias, _Float16* __restrict__ Out) {
  const int wave = threadIdx.x >> 5;
  const int lane = threadIdx.x & 31, l16 = lane & 15, hf = lane >> 4;
  const int tile = blockIdx.x * 8 + wave;
  const int tm = tile >> 6;  // 64 N-tiles
  const int tn = tile & 63;
  const int row0 = tm * 16, col0 = tn * 16;

  v8f acc = {};
  for (int k0 = 0; k0 < DMODEL; k0 += 32) {
    v16h a = fragA_f32(X, DMODEL, row0, k0);
    v16h b = fragB_f32(W, DMODEL, col0, k0);
    acc = wmma_f16(a, b, acc);
  }

  const int col = col0 + l16;
  const int h = col >> 6, d = col & 63;
  const float bz = bias[col];
  const int bb = row0 >> 11;           // all 16 rows share the batch index
  const int q0 = row0 & (NQ - 1);

  if constexpr (TRANSPOSED) {
    v8h pack;
#pragma unroll
    for (int r = 0; r < 8; ++r) pack[r] = (_Float16)(acc[r] + bz);
    _Float16* p =
        Out + (((size_t)(bb * NHEAD + h)) * DK + d) * NQ + (q0 + hf * 8);
    *(v8h*)p = pack;                   // 16B-aligned: q0+hf*8 is 8-aligned
  } else {
#pragma unroll
    for (int r = 0; r < 8; ++r) {
      const int q = q0 + r + hf * 8;
      Out[(((size_t)(bb * NHEAD + h)) * NQ + q) * DK + d] =
          (_Float16)(acc[r] + bz);
    }
  }
}

// ---------------------------------------------------------------------------
// Kernel 2: flash attention. One wave = one (b,h,16-query tile); streams all
// 2048 keys in 32-key blocks with online softmax. attw/mask reads (0.67 GB,
// read-once) use non-temporal loads + prefetch so the reused f16 Q/K/V^T
// operands stay resident in the 192 MB L2.
// ---------------------------------------------------------------------------
__global__ __launch_bounds__(128) void attn_kernel(
    const _Float16* __restrict__ Qh, const _Float16* __restrict__ Kh,
    const _Float16* __restrict__ VTh, const float* __restrict__ AW,
    const unsigned char* __restrict__ MK, _Float16* __restrict__ Oh) {
  __shared__ _Float16 Pbuf[4][16 * 32];  // per-wave P staging (C->A layout)

  const int wave = threadIdx.x >> 5;
  const int lane = threadIdx.x & 31, l16 = lane & 15, hf = lane >> 4;
  const int wid = blockIdx.x * 4 + wave;
  const int qt = wid & (NQ / 16 - 1);    // 128 q-tiles
  const int bh = wid >> 7;               // 0..31 = b*16+h
  const size_t head = (size_t)bh * NQ * DK;  // also valid for VT (64*NQ)
  const size_t headAW = (size_t)bh * NQ * (size_t)NK;
  const int q0 = qt * 16;

  // Q fragments are invariant over the key loop
  const v16h aq0 = fragA_f16(Qh + head, DK, q0, 0);
  const v16h aq1 = fragA_f16(Qh + head, DK, q0, 32);

  float rmax[8], rsum[8];
#pragma unroll
  for (int r = 0; r < 8; ++r) { rmax[r] = -1e30f; rsum[r] = 0.0f; }
  v8f acc0 = {}, acc1 = {}, acc2 = {}, acc3 = {};

  const float scale = 0.125f;  // 1/sqrt(64)
  _Float16* pb = Pbuf[wave];

  for (int kb = 0; kb < NK; kb += 32) {
    // S = Q(16x64) @ K^T(64x32): two 16-col tiles, two 32-deep k-steps each
    v8f s0 = {}, s1 = {};
    s0 = wmma_f16(aq0, fragB_f16(Kh + head, DK, kb, 0), s0);
    s0 = wmma_f16(aq1, fragB_f16(Kh + head, DK, kb, 32), s0);
    s1 = wmma_f16(aq0, fragB_f16(Kh + head, DK, kb + 16, 0), s1);
    s1 = wmma_f16(aq1, fragB_f16(Kh + head, DK, kb + 16, 32), s1);

    float p0[8], p1[8];
#pragma unroll
    for (int r = 0; r < 8; ++r) {
      const size_t rowo = headAW + (size_t)(q0 + r + hf * 8) * NK + kb;
      // streaming, read-once tensors: NT loads + prefetch next key-block
      const float aw0 = __builtin_nontemporal_load(&AW[rowo + l16]);
      const float aw1 = __builtin_nontemporal_load(&AW[rowo + 16 + l16]);
      const unsigned int mk0 = __builtin_nontemporal_load(&MK[rowo + l16]);
      const unsigned int mk1 = __builtin_nontemporal_load(&MK[rowo + 16 + l16]);
      __builtin_prefetch(&AW[rowo + 32], 0, 0);   // global_prefetch_b8
      __builtin_prefetch(&MK[rowo + 32], 0, 0);
      float v0 = s0[r] * scale * aw0;
      float v1 = s1[r] * scale * aw1;
      if (mk0) v0 = -1e30f;  // mask -> -inf (finite to avoid NaN)
      if (mk1) v1 = -1e30f;
      const float mn = fmaxf(rmax[r], half16_max(fmaxf(v0, v1)));
      const float fr = __expf(rmax[r] - mn);
      rmax[r] = mn;
      const float e0 = __expf(v0 - mn);
      const float e1 = __expf(v1 - mn);
      rsum[r] = rsum[r] * fr + half16_sum(e0 + e1);
      acc0[r] *= fr; acc1[r] *= fr; acc2[r] *= fr; acc3[r] *= fr;
      p0[r] = e0; p1[r] = e1;
    }

    // C-layout -> A-layout transpose of P(16x32) through LDS (per-wave; LDS
    // ops are processed in-order within a wave)
#pragma unroll
    for (int r = 0; r < 8; ++r) {
      pb[(r + hf * 8) * 32 + l16]      = (_Float16)p0[r];
      pb[(r + hf * 8) * 32 + 16 + l16] = (_Float16)p1[r];
    }
    v16h pa;
#pragma unroll
    for (int i = 0; i < 8; ++i) {
      pa[i]     = pb[l16 * 32 + hf * 8 + i];
      pa[i + 8] = pb[l16 * 32 + 16 + hf * 8 + i];
    }

    // O += P(16x32) @ V(32x64); V^T storage makes B-columns contiguous in K
    acc0 = wmma_f16(pa, fragB_f16(VTh + head, NQ, 0,  kb), acc0);
    acc1 = wmma_f16(pa, fragB_f16(VTh + head, NQ, 16, kb), acc1);
    acc2 = wmma_f16(pa, fragB_f16(VTh + head, NQ, 32, kb), acc2);
    acc3 = wmma_f16(pa, fragB_f16(VTh + head, NQ, 48, kb), acc3);
  }

  // normalize and store O as f16 [B,H,NQ,64]
#pragma unroll
  for (int r = 0; r < 8; ++r) {
    const float inv = 1.0f / rsum[r];
    const size_t rowo = head + (size_t)(q0 + r + hf * 8) * DK;
    Oh[rowo + l16]      = (_Float16)(acc0[r] * inv);
    Oh[rowo + 16 + l16] = (_Float16)(acc1[r] * inv);
    Oh[rowo + 32 + l16] = (_Float16)(acc2[r] * inv);
    Oh[rowo + 48 + l16] = (_Float16)(acc3[r] * inv);
  }
}

// ---------------------------------------------------------------------------
// Kernel 3: output projection out[4096,1024] = O @ Wo^T + bo (f32 out).
// A logical col c = h*64+d maps into the [B,H,NQ,64] O buffer; 32-wide
// K-chunks never straddle a head boundary. Output is write-once -> NT store.
// ---------------------------------------------------------------------------
__global__ __launch_bounds__(256) void oproj_kernel(
    const _Float16* __restrict__ Oh, const float* __restrict__ Wo,
    const float* __restrict__ bo, float* __restrict__ out) {
  const int wave = threadIdx.x >> 5;
  const int lane = threadIdx.x & 31, l16 = lane & 15, hf = lane >> 4;
  const int tile = blockIdx.x * 8 + wave;
  const int tm = tile >> 6, tn = tile & 63;
  const int row0 = tm * 16, col0 = tn * 16;
  const int bb = row0 >> 11, q0 = row0 & (NQ - 1);

  v8f acc = {};
  for (int c0 = 0; c0 < DMODEL; c0 += 32) {
    const int hh = c0 >> 6, d0 = c0 & 63;
    v16h a = fragA_f16(Oh + ((size_t)(bb * NHEAD + hh)) * NQ * DK, DK, q0, d0);
    v16h b = fragB_f32(Wo, DMODEL, col0, c0);
    acc = wmma_f16(a, b, acc);
  }

  const float bz = bo[col0 + l16];
#pragma unroll
  for (int r = 0; r < 8; ++r) {
    const size_t grow = (size_t)(row0 + r + hf * 8);
    __builtin_nontemporal_store(acc[r] + bz, &out[grow * DMODEL + col0 + l16]);
  }
}

// ---------------------------------------------------------------------------
extern "C" void kernel_launch(void* const* d_in, const int* in_sizes, int n_in,
                              void* d_out, int out_size, void* d_ws,
                              size_t ws_size, hipStream_t stream) {
  (void)in_sizes; (void)n_in; (void)out_size; (void)ws_size;
  const float* queries = (const float*)d_in[0];
  const float* keys    = (const float*)d_in[1];
  const float* values  = (const float*)d_in[2];
  const unsigned char* mask = (const unsigned char*)d_in[3];  // bool
  const float* attw = (const float*)d_in[4];
  const float* Wq = (const float*)d_in[5];
  const float* bq = (const float*)d_in[6];
  const float* Wk = (const float*)d_in[7];
  const float* bk = (const float*)d_in[8];
  const float* Wv = (const float*)d_in[9];
  const float* bv = (const float*)d_in[10];
  const float* Wo = (const float*)d_in[11];
  const float* bo = (const float*)d_in[12];

  const size_t QSZ = (size_t)NBATCH * NHEAD * NQ * DK;  // 4M f16 = 8 MB
  _Float16* Qh  = (_Float16*)d_ws;
  _Float16* Kh  = Qh + QSZ;
  _Float16* VTh = Kh + QSZ;
  _Float16* Oh  = VTh + QSZ;

  // 4096x1024 output tiles: 256*64 tiles / 8 waves per block = 2048 blocks
  proj_kernel<false><<<2048, 256, 0, stream>>>(queries, Wq, bq, Qh);
  proj_kernel<false><<<2048, 256, 0, stream>>>(keys, Wk, bk, Kh);
  proj_kernel<true><<<2048, 256, 0, stream>>>(values, Wv, bv, VTh);

  // 2*16*128 q-tiles / 4 waves per block = 1024 blocks
  attn_kernel<<<1024, 128, 0, stream>>>(Qh, Kh, VTh, attw, mask, Oh);

  oproj_kernel<<<2048, 256, 0, stream>>>(Oh, Wo, bo, (float*)d_out);
}
